// Text_Concat_Vision_33681133535327
// MI455X (gfx1250) — matrix operand order repaired and checked by reading, not compiled, not run on hardware
//
#include <hip/hip_runtime.h>

typedef __bf16 bh16;
typedef __attribute__((ext_vector_type(16))) __bf16 v16bh;
typedef __attribute__((ext_vector_type(8)))  __bf16 v8bh;
typedef __attribute__((ext_vector_type(8)))  float  v8f;

#define EPSV 1e-8f
#define FMAXV 3.402823466e+38f

// ---------------- conversion / padding kernels ----------------

__global__ void k_cvt_bf16(const float* __restrict__ src, bh16* __restrict__ dst, size_t n) {
  size_t i = (size_t)blockIdx.x * blockDim.x + threadIdx.x;
  if (i < n) dst[i] = (bh16)src[i];
}

// dst [dr, dc] <- src [sr, sc], zero-padded outside
__global__ void k_pad_cvt(const float* __restrict__ src, bh16* __restrict__ dst,
                          int sr, int sc, int dr, int dc) {
  size_t i = (size_t)blockIdx.x * blockDim.x + threadIdx.x;
  size_t total = (size_t)dr * dc;
  if (i >= total) return;
  int r = (int)(i / dc), c = (int)(i % dc);
  float v = (r < sr && c < sc) ? src[(size_t)r * sc + c] : 0.f;
  dst[i] = (bh16)v;
}

__global__ void k_pad_bias(const float* __restrict__ src, float* __restrict__ dst, int sn, int dn) {
  int i = blockIdx.x * blockDim.x + threadIdx.x;
  if (i < dn) dst[i] = (i < sn) ? src[i] : 0.f;
}

// ---------------- WMMA bf16 GEMM:  C = relu(A @ W^T + bias) ----------------
// A: [M, K] bf16 row-major (stride K).  W: [N, K] bf16 row-major (stride K).
// Async (TDM-class) global->LDS staging, double-buffered, software pipelined.

template <int BM, int BN, int WAVES_M, int WAVES_N, bool OUT_BF16>
__global__ __launch_bounds__(256)
void gemm_bf16_relu(const bh16* __restrict__ A, const bh16* __restrict__ W,
                    const float* __restrict__ bias, void* __restrict__ Cout,
                    int N, int K) {
  constexpr int WM = BM / WAVES_M;   // wave tile M
  constexpr int WN = BN / WAVES_N;   // wave tile N
  constexpr int MF = WM / 16;
  constexpr int NF = WN / 16;
  constexpr int ST = 40;             // LDS row stride in bh16 (32 + 8 pad, 80B)
  constexpr int LA = (BM * 4) / 256;            // 16B A-chunks per thread (2 or 4)
  constexpr int LB = (BN * 4 + 255) / 256;      // 16B B-chunks per thread (dup if BN*4<256)
  constexpr unsigned ABUF = BM * ST * 2;        // bytes per A buffer
  constexpr unsigned BBUF = BN * ST * 2;        // bytes per B buffer

  __shared__ __align__(16) bh16 sA[2][BM * ST];
  __shared__ __align__(16) bh16 sB[2][BN * ST];

  const int tid  = threadIdx.x;
  const int wave = tid >> 5;
  const int lane = tid & 31;
  const int r    = lane & 15;
  const int hi   = lane >> 4;
  const int wm   = wave / WAVES_N;
  const int wn   = wave % WAVES_N;
  const int m0   = blockIdx.x * BM;
  const int n0   = blockIdx.y * BN;

  // ---- per-thread async staging addresses (bumped 64B per K-step) ----
  unsigned long long ga[LA], gb[LB];
  unsigned la[LA], lb[LB];
#pragma unroll
  for (int i = 0; i < LA; i++) {
    int c = tid + 256 * i;
    int row = c >> 2, c8 = (c & 3) << 3;
    ga[i] = (unsigned long long)(const void*)(A + (size_t)(m0 + row) * K + c8);
    la[i] = (unsigned)(uintptr_t)&sA[0][row * ST + c8];
  }
#pragma unroll
  for (int j = 0; j < LB; j++) {
    int c = (BN * 4 >= 256) ? (tid + 256 * j) : (tid & (BN * 4 - 1));
    int row = c >> 2, c8 = (c & 3) << 3;
    gb[j] = (unsigned long long)(const void*)(W + (size_t)(n0 + row) * K + c8);
    lb[j] = (unsigned)(uintptr_t)&sB[0][row * ST + c8];
  }

  v8f acc[MF][NF];
#pragma unroll
  for (int i = 0; i < MF; i++)
#pragma unroll
    for (int j = 0; j < NF; j++)
#pragma unroll
      for (int e = 0; e < 8; e++) acc[i][j][e] = 0.f;

  // ---- prologue: async-stage tile 0 into buffer 0 ----
#pragma unroll
  for (int i = 0; i < LA; i++)
    asm volatile("global_load_async_to_lds_b128 %0, %1, off"
                 :: "v"(la[i]), "v"(ga[i]) : "memory");
#pragma unroll
  for (int j = 0; j < LB; j++)
    asm volatile("global_load_async_to_lds_b128 %0, %1, off"
                 :: "v"(lb[j]), "v"(gb[j]) : "memory");
#pragma unroll
  for (int i = 0; i < LA; i++) ga[i] += 64;
#pragma unroll
  for (int j = 0; j < LB; j++) gb[j] += 64;

  const int nsteps = K >> 5;
  for (int s = 0; s < nsteps; s++) {
    const int buf = s & 1;
    // my in-flight tile-s loads complete; barrier => everyone's complete,
    // and everyone is done reading the other buffer (prev iteration finished)
    asm volatile("s_wait_asynccnt 0x0" ::: "memory");
    __syncthreads();

    if (s + 1 < nsteps) {
      const unsigned boff = buf ? 0u : 1u;  // next buffer
#pragma unroll
      for (int i = 0; i < LA; i++)
        asm volatile("global_load_async_to_lds_b128 %0, %1, off"
                     :: "v"(la[i] + boff * ABUF), "v"(ga[i]) : "memory");
#pragma unroll
      for (int j = 0; j < LB; j++)
        asm volatile("global_load_async_to_lds_b128 %0, %1, off"
                     :: "v"(lb[j] + boff * BBUF), "v"(gb[j]) : "memory");
#pragma unroll
      for (int i = 0; i < LA; i++) ga[i] += 64;
#pragma unroll
      for (int j = 0; j < LB; j++) gb[j] += 64;
    }

    // A fragment (16x32 MxK, 16-bit): lane<16 K={0..7,16..23}, lane>=16 K={8..15,24..31}
    v16bh af[MF];
#pragma unroll
    for (int i = 0; i < MF; i++) {
      int row = wm * WM + i * 16 + r;
      union { v16bh v; v8bh h[2]; } u;
      u.h[0] = *(const v8bh*)&sA[buf][row * ST + hi * 8];
      u.h[1] = *(const v8bh*)&sA[buf][row * ST + 16 + hi * 8];
      af[i] = u.v;
    }
    // B fragment (32x16 KxN): lane<16 K={0..7,16..23} of col r, lane>=16 K={8..15,24..31}
    v16bh bfrag[NF];
#pragma unroll
    for (int j = 0; j < NF; j++) {
      int row = wn * WN + j * 16 + r;   // row of W == column of B
      union { v16bh v; v8bh h[2]; } u;
      u.h[0] = *(const v8bh*)&sB[buf][row * ST + hi * 8];
      u.h[1] = *(const v8bh*)&sB[buf][row * ST + 16 + hi * 8];
      bfrag[j] = u.v;
    }
#pragma unroll
    for (int i = 0; i < MF; i++)
#pragma unroll
      for (int j = 0; j < NF; j++)
        acc[i][j] = __builtin_amdgcn_wmma_f32_16x16x32_bf16(
            false, af[i], false, bfrag[j], (short)0, acc[i][j], false, false);
  }

  // epilogue: bias + relu, per-lane stores (D layout: n=lane&15, m=vgpr+8*hi)
#pragma unroll
  for (int j = 0; j < NF; j++) {
    int col = n0 + wn * WN + j * 16 + r;
    float bb = bias[col];
#pragma unroll
    for (int i = 0; i < MF; i++) {
#pragma unroll
      for (int e = 0; e < 8; e++) {
        int rowm = m0 + wm * WM + i * 16 + hi * 8 + e;
        float v = acc[i][j][e] + bb;
        v = v > 0.f ? v : 0.f;
        if constexpr (OUT_BF16)
          ((bh16*)Cout)[(size_t)rowm * N + col] = (bh16)v;
        else
          ((float*)Cout)[(size_t)rowm * N + col] = v;
      }
    }
  }
}

// ---------------- fused tail: cov -> fused -> alpha/mu/logvar -> z -> pred ----------------

__device__ inline float wave_sum(float v) {
#pragma unroll
  for (int off = 16; off > 0; off >>= 1) v += __shfl_xor(v, off, 32);
  return v;
}
__device__ inline float wave_max(float v) {
#pragma unroll
  for (int off = 16; off > 0; off >>= 1) v = fmaxf(v, __shfl_xor(v, off, 32));
  return v;
}
__device__ inline float block_sum(float v, float* s4) {
  v = wave_sum(v);
  if ((threadIdx.x & 31) == 0) s4[threadIdx.x >> 5] = v;
  __syncthreads();
  v = s4[0] + s4[1] + s4[2] + s4[3];
  __syncthreads();
  return v;
}
__device__ inline float block_max(float v, float* s4) {
  v = wave_max(v);
  if ((threadIdx.x & 31) == 0) s4[threadIdx.x >> 5] = v;
  __syncthreads();
  v = fmaxf(fmaxf(s4[0], s4[1]), fmaxf(s4[2], s4[3]));
  __syncthreads();
  return v;
}

__global__ __launch_bounds__(128)
void k_tail(const float* __restrict__ t32, const float* __restrict__ v32,
            const float* __restrict__ epsin,
            const float* __restrict__ fus_w, const float* __restrict__ fus_b,
            const float* __restrict__ al_w, const float* __restrict__ al_b,
            const float* __restrict__ mu_w, const float* __restrict__ mu_b,
            const float* __restrict__ lv_w, const float* __restrict__ lv_b,
            const float* __restrict__ fc_w, const float* __restrict__ fc_b,
            float* __restrict__ pred, float* __restrict__ cov_out) {
  __shared__ float s_cov[32];
  __shared__ float s_f[128];
  __shared__ float s_p[128];
  __shared__ float s4a[4], s4b[4];
  const int row = blockIdx.x;
  const int j = threadIdx.x;

  // ---- beta-liouville covariance (wave 0 only; 32 lanes fully active) ----
  if (j < 32) {
    float t = t32[(size_t)row * 32 + j];
    float v = v32[(size_t)row * 32 + j];
    float nt = sqrtf(wave_sum(t * t)) + EPSV;
    float nv = sqrtf(wave_sum(v * v)) + EPSV;
    float a1 = t / nt, a2 = v / nv;
    float asum = a1 + a2 + EPSV;
    float term1 = (a1 + 1.f) / (asum + 1.f);
    float term2 = a1 / asum;
    float f1 = term1 * term2 / (asum + 1.f);
    float f2 = term2 / asum;
    float cov = (a1 * a2 / asum) * (f1 - f2);
    if (cov != cov) cov = 0.f;                 // nan_to_num
    else if (cov > FMAXV) cov = FMAXV;
    else if (cov < -FMAXV) cov = -FMAXV;
    cov_out[(size_t)row * 32 + j] = cov;
    s_cov[j] = cov;
  }
  __syncthreads();

  // ---- fused_ = relu(cov @ fus_w.T + fus_b), 128 outputs ----
  float f = fus_b[j];
#pragma unroll 8
  for (int k = 0; k < 32; k++) f += s_cov[k] * fus_w[j * 32 + k];
  f = fmaxf(f, 0.f);
  s_f[j] = f;
  __syncthreads();

  // ---- alpha / mu / logvar (three 128x128 layers) ----
  float a = al_b[j], m = mu_b[j], lv = lv_b[j];
  for (int k = 0; k < 128; k++) {
    float fk = s_f[k];
    a  += fk * al_w[j * 128 + k];
    m  += fk * mu_w[j * 128 + k];
    lv += fk * lv_w[j * 128 + k];
  }
  float as = epsin[(size_t)row * 128 + j] * expf(lv) + m + a;

  // ---- softmax over 128 ----
  float mx = block_max(as, s4a);
  float e = expf(as - mx);
  float se = block_sum(e, s4b);
  float rho = e / se;

  // ---- stick-breaking: exclusive multiplicative scan of (1 - rho) ----
  s_p[j] = 1.f - rho;
  __syncthreads();
  for (int off = 1; off < 128; off <<= 1) {
    float x = s_p[j];
    float y = (j >= off) ? s_p[j - off] : 1.f;
    __syncthreads();
    s_p[j] = x * y;
    __syncthreads();
  }
  float excl = (j == 0) ? 1.f : s_p[j - 1];
  float z = rho * excl;

  // ---- pred = sigmoid(z . fc_w + fc_b) ----
  float s = block_sum(z * fc_w[j], s4a);
  if (j == 0) pred[row] = 1.f / (1.f + expf(-(s + fc_b[0])));
}

// ---------------- host launch ----------------

extern "C" void kernel_launch(void* const* d_in, const int* in_sizes, int n_in,
                              void* d_out, int out_size, void* d_ws, size_t ws_size,
                              hipStream_t stream) {
  (void)in_sizes; (void)n_in; (void)out_size; (void)ws_size;
  constexpr int Bn = 16384, TD = 768, VD = 4096, F1 = 2742, NP = 2816, F2 = 32;

  const float* bert  = (const float*)d_in[0];
  const float* vgg   = (const float*)d_in[1];
  const float* epsin = (const float*)d_in[2];
  const float* tw1   = (const float*)d_in[3];
  const float* tb1   = (const float*)d_in[4];
  const float* tw2   = (const float*)d_in[5];
  const float* tb2   = (const float*)d_in[6];
  const float* vw1   = (const float*)d_in[7];
  const float* vb1   = (const float*)d_in[8];
  const float* vw2   = (const float*)d_in[9];
  const float* vb2   = (const float*)d_in[10];
  const float* fus_w = (const float*)d_in[11];
  const float* fus_b = (const float*)d_in[12];
  const float* al_w  = (const float*)d_in[13];
  const float* al_b  = (const float*)d_in[14];
  const float* mu_w  = (const float*)d_in[15];
  const float* mu_b  = (const float*)d_in[16];
  const float* lv_w  = (const float*)d_in[17];
  const float* lv_b  = (const float*)d_in[18];
  const float* fc_w  = (const float*)d_in[19];
  const float* fc_b  = (const float*)d_in[20];

  float* pred    = (float*)d_out;
  float* cov_out = pred + Bn;

  char* ws = (char*)d_ws;
  size_t off = 0;
  auto take = [&](size_t bytes) -> void* {
    void* p = ws + off;
    off += (bytes + 255) & ~(size_t)255;
    return p;
  };
  bh16* bertbf = (bh16*)take((size_t)Bn * TD * 2);
  bh16* vggbf  = (bh16*)take((size_t)Bn * VD * 2);
  bh16* tw1p   = (bh16*)take((size_t)NP * TD * 2);
  bh16* vw1p   = (bh16*)take((size_t)NP * VD * 2);
  bh16* tw2p   = (bh16*)take((size_t)F2 * NP * 2);
  bh16* vw2p   = (bh16*)take((size_t)F2 * NP * 2);
  float* tb1p  = (float*)take((size_t)NP * 4);
  float* vb1p  = (float*)take((size_t)NP * 4);
  bh16* th     = (bh16*)take((size_t)Bn * NP * 2);
  bh16* vh     = (bh16*)take((size_t)Bn * NP * 2);
  float* t32   = (float*)take((size_t)Bn * F2 * 4);
  float* v32   = (float*)take((size_t)Bn * F2 * 4);

  // --- convert / pad ---
  {
    size_t n = (size_t)Bn * TD;
    k_cvt_bf16<<<dim3((unsigned)((n + 255) / 256)), 256, 0, stream>>>(bert, bertbf, n);
    n = (size_t)Bn * VD;
    k_cvt_bf16<<<dim3((unsigned)((n + 255) / 256)), 256, 0, stream>>>(vgg, vggbf, n);
  }
  k_pad_cvt<<<dim3((unsigned)(((size_t)NP * TD + 255) / 256)), 256, 0, stream>>>(tw1, tw1p, F1, TD, NP, TD);
  k_pad_cvt<<<dim3((unsigned)(((size_t)NP * VD + 255) / 256)), 256, 0, stream>>>(vw1, vw1p, F1, VD, NP, VD);
  k_pad_cvt<<<dim3((unsigned)(((size_t)F2 * NP + 255) / 256)), 256, 0, stream>>>(tw2, tw2p, F2, F1, F2, NP);
  k_pad_cvt<<<dim3((unsigned)(((size_t)F2 * NP + 255) / 256)), 256, 0, stream>>>(vw2, vw2p, F2, F1, F2, NP);
  k_pad_bias<<<dim3((NP + 255) / 256), 256, 0, stream>>>(tb1, tb1p, F1, NP);
  k_pad_bias<<<dim3((NP + 255) / 256), 256, 0, stream>>>(vb1, vb1p, F1, NP);

  // --- layer 1 GEMMs: [16384, K] x [K, 2816] -> bf16 hidden ---
  dim3 g1(Bn / 128, NP / 128);
  gemm_bf16_relu<128, 128, 2, 4, true><<<g1, 256, 0, stream>>>(bertbf, tw1p, tb1p, th, NP, TD);
  gemm_bf16_relu<128, 128, 2, 4, true><<<g1, 256, 0, stream>>>(vggbf, vw1p, vb1p, vh, NP, VD);

  // --- layer 2 GEMMs: [16384, 2816] x [2816, 32] -> f32 ---
  dim3 g2(Bn / 256, 1);
  gemm_bf16_relu<256, 32, 8, 1, false><<<g2, 256, 0, stream>>>(th, tw2p, tb2, t32, F2, NP);
  gemm_bf16_relu<256, 32, 8, 1, false><<<g2, 256, 0, stream>>>(vh, vw2p, vb2, v32, F2, NP);

  // --- fused tail: one 128-thread block per batch row ---
  k_tail<<<dim3(Bn), 128, 0, stream>>>(t32, v32, epsin,
                                       fus_w, fus_b, al_w, al_b, mu_w, mu_b,
                                       lv_w, lv_b, fc_w, fc_b, pred, cov_out);
}
